// FastDTWEmbeddingSimilarity_35639638622827
// MI455X (gfx1250) — matrix operand order repaired and checked by reading, not compiled, not run on hardware
//
#include <hip/hip_runtime.h>
#include <hip/hip_bf16.h>
#include <math.h>

// ---------------------------------------------------------------------------
// FastDTW embedding similarity for MI455X (gfx1250, wave32, WMMA).
//
// Pipeline (all on `stream`, sequential dependencies via stream order):
//   1) normalize_rows : f32 [16,1024,768] -> row-L2-normalized f16 in ws
//   2) cosine_gemm    : dist[b] = 1 - A[b] (1024x768) * B[b]^T  via
//                       v_wmma_f32_16x16x32_f16, f32 accum, 2x2 tile/wave
//   3) dtw_tiled      : 32x32-tile anti-diagonal DP, one wave per tile,
//                       intra-tile via wave32 shuffles, 63 barriers/batch
//
// Workspace layout (d_ws):
//   [0,            25165824)  A  f16  16*1024*768
//   [25165824,     50331648)  B  f16  16*1024*768
//   [50331648,    117440512)  dist f32 16*1024*1024
// ---------------------------------------------------------------------------

typedef __attribute__((ext_vector_type(16))) _Float16 v16h;
typedef __attribute__((ext_vector_type(8)))  _Float16 v8h;
typedef __attribute__((ext_vector_type(8)))  float    v8f;

#define SEQ 1024
#define DIM 768
#define NB  16
#define TILE 32
#define NTL (SEQ / TILE)          // 32 tiles per dimension
#define NDIAG (2 * TILE - 1)      // 63 intra-tile anti-diagonals

// ---------------------------------------------------------------------------
// 1) Row normalization: one 256-thread block per row (2*16*1024 rows total).
//    scale = 1 / max(||x||2, 1e-12), matching torch F.normalize eps.
// ---------------------------------------------------------------------------
__global__ void normalize_rows_kernel(const float* __restrict__ s1,
                                      const float* __restrict__ s2,
                                      _Float16* __restrict__ A,
                                      _Float16* __restrict__ B) {
    int r = blockIdx.x;                       // 0 .. 2*NB*SEQ-1
    const float* src;
    _Float16*    dst;
    if (r < NB * SEQ) {
        src = s1 + (size_t)r * DIM;
        dst = A  + (size_t)r * DIM;
    } else {
        int q = r - NB * SEQ;
        src = s2 + (size_t)q * DIM;
        dst = B  + (size_t)q * DIM;
    }

    __shared__ float red[256];
    float s = 0.0f;
    for (int i = threadIdx.x; i < DIM; i += 256) {
        float v = src[i];
        s += v * v;
    }
    red[threadIdx.x] = s;
    __syncthreads();
    for (int off = 128; off > 0; off >>= 1) {
        if (threadIdx.x < off) red[threadIdx.x] += red[threadIdx.x + off];
        __syncthreads();
    }
    float norm  = sqrtf(red[0]);
    float scale = 1.0f / fmaxf(norm, 1e-12f);
    for (int i = threadIdx.x; i < DIM; i += 256) {
        dst[i] = (_Float16)(src[i] * scale);
    }
}

// ---------------------------------------------------------------------------
// 2) Cosine-distance GEMM with WMMA.
//
// CDNA5 V_WMMA_F32_16X16X32_F16 operand layouts (ISA 7.12.2) map directly to
// row-major [seq,dim] storage of BOTH operands of A*B^T:
//   A frag (16x32, MxK): lane L<16 -> row M=L, elems 0..7 = K[k0..k0+7],
//                        elems 8..15 = K[k0+16..k0+23]; lanes 16..31 shift K by 8.
//   B frag (32x16, KxN): B[k][n] = s2n[n][k]; lane L<16 -> col N=L,
//                        elems 0..7 = K[k0..k0+7], 8..15 = K[k0+16..k0+23]; lanes 16..31 K += 8.
//   C/D (16x16 f32):     lane L<16: VGPR v -> (M=v, N=L); lanes 16..31: M=8+v.
// Each wave computes a 32x32 output tile (2x2 fragments) reusing A/B frags.
// ---------------------------------------------------------------------------
__device__ __forceinline__ v16h load_a_frag(const _Float16* __restrict__ row,
                                            int k0, int hi) {
    v8h lo = *(const v8h*)(row + k0 + 8 * hi);
    v8h hh = *(const v8h*)(row + k0 + 16 + 8 * hi);
    return __builtin_shufflevector(lo, hh, 0, 1, 2, 3, 4, 5, 6, 7,
                                           8, 9, 10, 11, 12, 13, 14, 15);
}

__device__ __forceinline__ v16h load_b_frag(const _Float16* __restrict__ row,
                                            int k0, int hi) {
    v8h lo = *(const v8h*)(row + k0 + 8 * hi);
    v8h hh = *(const v8h*)(row + k0 + 16 + 8 * hi);
    return __builtin_shufflevector(lo, hh, 0, 1, 2, 3, 4, 5, 6, 7,
                                           8, 9, 10, 11, 12, 13, 14, 15);
}

__device__ __forceinline__ void store_dist_tile(float* __restrict__ dbase,
                                                const v8f& c, int m0, int n,
                                                int hi) {
#pragma unroll
    for (int v = 0; v < 8; ++v) {
        int m = m0 + 8 * hi + v;
        dbase[(size_t)m * SEQ + n] = 1.0f - c[v];
    }
}

__global__ void cosine_gemm_kernel(const _Float16* __restrict__ A,
                                   const _Float16* __restrict__ B,
                                   float* __restrict__ dist) {
    const int lane    = threadIdx.x & 31;
    const int wave    = threadIdx.x >> 5;              // 8 waves / block
    const int t       = blockIdx.x * 8 + wave;         // wave tile id
    const int nTiles  = NB * (SEQ / 32) * (SEQ / 32);  // 16384
    if (t >= nTiles) return;

    const int b   = t >> 10;
    const int rem = t & 1023;
    const int mo  = (rem >> 5) << 5;
    const int no  = (rem & 31) << 5;

    const int laneLow = lane & 15;
    const int hi      = lane >> 4;

    const _Float16* aRow0 = A + ((size_t)(b * SEQ + mo + laneLow)) * DIM;
    const _Float16* aRow1 = aRow0 + (size_t)16 * DIM;
    const _Float16* bRow0 = B + ((size_t)(b * SEQ + no + laneLow)) * DIM;
    const _Float16* bRow1 = bRow0 + (size_t)16 * DIM;

    v8f c00 = {}, c01 = {}, c10 = {}, c11 = {};

#pragma unroll 4
    for (int k0 = 0; k0 < DIM; k0 += 32) {
        v16h a0 = load_a_frag(aRow0, k0, hi);
        v16h a1 = load_a_frag(aRow1, k0, hi);
        v16h b0 = load_b_frag(bRow0, k0, hi);
        v16h b1 = load_b_frag(bRow1, k0, hi);
        c00 = __builtin_amdgcn_wmma_f32_16x16x32_f16(false, a0, false, b0,
                                                     (short)0, c00, false, false);
        c01 = __builtin_amdgcn_wmma_f32_16x16x32_f16(false, a0, false, b1,
                                                     (short)0, c01, false, false);
        c10 = __builtin_amdgcn_wmma_f32_16x16x32_f16(false, a1, false, b0,
                                                     (short)0, c10, false, false);
        c11 = __builtin_amdgcn_wmma_f32_16x16x32_f16(false, a1, false, b1,
                                                     (short)0, c11, false, false);
    }

    float* dbase = dist + ((size_t)b << 20);
    store_dist_tile(dbase, c00, mo,      no + laneLow,      hi);
    store_dist_tile(dbase, c01, mo,      no + 16 + laneLow, hi);
    store_dist_tile(dbase, c10, mo + 16, no + laneLow,      hi);
    store_dist_tile(dbase, c11, mo + 16, no + 16 + laneLow, hi);
}

// ---------------------------------------------------------------------------
// 3) Tiled DTW wavefront. One 1024-thread block (32 waves) per batch.
//    Tile grid 32x32; on tile-diagonal td, wave w owns tile (ti=td-w, tj=w).
//    Intra-tile: lane l = local column; skewed step s computes cell
//    (r=s-l, l). Recurrence:
//      val = dist + min(top, left, diag);  top  = own prev-step value,
//      left = shfl_up(prev, 1),            diag = shfl_up(prev2, 1).
//    Tile boundary values flow through 3-way (mod td) rotated LDS buffers:
//      rowBuf = bottom rows of tiles (read by tile below, td-1; corner td-2)
//      colBuf = right columns of tiles (read by tile to the right, td-1)
//    All LDS/global reads are hoisted out of the 63-step dependency chain:
//    dist anti-diagonals preloaded to a statically-indexed VGPR array
//    (slot index s is loop-uniform under full unroll), left-boundary column
//    preloaded per lane and delivered per step via uniform __shfl.
// ---------------------------------------------------------------------------
__global__ __launch_bounds__(1024) void dtw_tiled_kernel(
        const float* __restrict__ dist, float* __restrict__ out) {
    const int b = blockIdx.x;
    const float* __restrict__ D = dist + ((size_t)b << 20);

    __shared__ float rowBuf[3][SEQ];
    __shared__ float colBuf[3][SEQ];

    const int   lane = threadIdx.x & 31;
    const int   tj   = threadIdx.x >> 5;     // wave id = tile column
    const float INF  = 3.0e38f;

    for (int td = 0; td < 2 * NTL - 1; ++td) {
        const int ti = td - tj;
        if (ti >= 0 && ti < NTL) {
            const int gr0 = ti * TILE;
            const int gc0 = tj * TILE;
            const int gc  = gc0 + lane;

            const float* rowPrev  = rowBuf[(td + 2) % 3];  // td-1
            const float* rowPrev2 = rowBuf[(td + 1) % 3];  // td-2
            const float* colPrev  = colBuf[(td + 2) % 3];  // td-1
            float* rowCur = rowBuf[td % 3];
            float* colCur = colBuf[td % 3];

            // ---- preload: dist anti-diagonals (off the dependency chain) --
            float dreg[NDIAG];
#pragma unroll
            for (int s = 0; s < NDIAG; ++s) {
                int r = s - lane;
                dreg[s] = (r >= 0 && r < TILE)
                            ? D[(size_t)(gr0 + r) * SEQ + gc] : 0.0f;
            }

            // ---- preload boundary values ---------------------------------
            // top boundary (bottom row of tile above) for this lane's column
            float rpv   = (ti > 0) ? rowPrev[gc] : INF;
            float rpvm1 = __shfl_up(rpv, 1, 32);          // rowPrev[gc-1]
            // left boundary (right col of left tile): lane l holds row gr0+l
            float cpv   = (tj > 0) ? colPrev[gr0 + lane] : INF;
            // corner dtw[gr0-1][gc0-1] from tile-diagonal td-2 (lane 0 only)
            float c00;
            if (ti > 0 && tj > 0)       c00 = rowPrev2[gc0 - 1];
            else if (ti == 0 && tj == 0) c00 = 0.0f;      // global cell (0,0)
            else                         c00 = INF;

            // ---- skewed wavefront: pure VALU + shuffles ------------------
            float prev = INF, prev2 = INF, cpv_prev = INF;
#pragma unroll
            for (int s = 0; s < NDIAG; ++s) {
                float leftVal = __shfl_up(prev, 1, 32);
                float diagVal = __shfl_up(prev2, 1, 32);
                float cpv_cur = __shfl(cpv, s, 32);       // colPrev[gr0+s]
                const int r   = s - lane;
                float val     = prev;                      // hold when idle
                if (r >= 0 && r < TILE) {
                    float top  = (r > 0)    ? prev    : rpv;
                    float left = (lane > 0) ? leftVal : cpv_cur;
                    float dg;
                    if (lane > 0) dg = (r > 0) ? diagVal  : rpvm1;
                    else          dg = (r > 0) ? cpv_prev : c00;
                    val = dreg[s] + fminf(fminf(top, left), dg);
                    if (r == TILE - 1)    rowCur[gc] = val;
                    if (lane == TILE - 1) colCur[gr0 + r] = val;
                    if (gr0 + r == SEQ - 1 && gc == SEQ - 1)
                        out[b] = 1.0f / (1.0f + val * (1.0f / (float)(2 * SEQ)));
                }
                prev2    = prev;
                prev     = val;
                cpv_prev = cpv_cur;
            }
        }
        __syncthreads();   // publish rowCur/colCur for tile-diagonal td+1
    }
}

// ---------------------------------------------------------------------------
// Launch
// ---------------------------------------------------------------------------
extern "C" void kernel_launch(void* const* d_in, const int* in_sizes, int n_in,
                              void* d_out, int out_size, void* d_ws, size_t ws_size,
                              hipStream_t stream) {
    const float* seq1 = (const float*)d_in[0];   // [16,1024,768] f32
    const float* seq2 = (const float*)d_in[1];   // [16,1024,768] f32
    float*       out  = (float*)d_out;           // [16] f32

    char* ws = (char*)d_ws;
    const size_t halfBytes = (size_t)NB * SEQ * DIM * sizeof(_Float16); // 25165824
    _Float16* A    = (_Float16*)ws;
    _Float16* B    = (_Float16*)(ws + halfBytes);
    float*    dist = (float*)(ws + 2 * halfBytes);                      // 64 MB

    normalize_rows_kernel<<<2 * NB * SEQ, 256, 0, stream>>>(seq1, seq2, A, B);

    cosine_gemm_kernel<<<(NB * (SEQ / 32) * (SEQ / 32)) / 8, 256, 0, stream>>>(A, B, dist);

    dtw_tiled_kernel<<<NB, SEQ, 0, stream>>>(dist, out);
}
